// EncoderRNN_6846177870456
// MI455X (gfx1250) — matrix operand order repaired and checked by reading, not compiled, not run on hardware
//
#include <hip/hip_runtime.h>
#include <hip/hip_bf16.h>
#include <math.h>

// Problem sizes (fixed by the reference)
#define VV 32000
#define HH 1024
#define LL 256
#define TT 4096
#define FH (4 * HH)   // 4096 gate rows

typedef __attribute__((ext_vector_type(16))) __bf16 v16bf;
typedef __attribute__((ext_vector_type(8)))  __bf16 v8bf;
typedef __attribute__((ext_vector_type(8)))  float  v8f;

// ---------- helpers ----------
static __device__ inline unsigned short f2bf_bits(float f) {
  unsigned u = __float_as_uint(f);
  u += 0x7fffu + ((u >> 16) & 1u);          // round-to-nearest-even
  return (unsigned short)(u >> 16);
}
static __device__ inline float bfbits2f(unsigned short h) {
  return __uint_as_float(((unsigned)h) << 16);
}
static __device__ inline float sigmoidf(float x) {
  return 1.0f / (1.0f + __expf(-x));
}
static __device__ inline float wave_reduce_add(float v) {
  #pragma unroll
  for (int off = 16; off > 0; off >>= 1) v += __shfl_xor(v, off, 32);
  return v;
}

// ---------- kernel 0: per-launch init (counter reset + h0 seed) ----------
__global__ void init_state(const float* __restrict__ h0, float* __restrict__ hbuf,
                           unsigned* __restrict__ counter) {
  int i = blockIdx.x * blockDim.x + threadIdx.x;
  if (i < HH) {
    hbuf[i]      = h0[i];   // buffer 0 = initial hidden
    hbuf[HH + i] = 0.0f;    // buffer 1 deterministic
  }
  if (i == 0) *counter = 0u;
}

// ---------- kernel 1a: w_hh -> bf16, reordered so each scan-WG reads contiguously ----------
// dst row layout: [wg][gate*16 + e][k],  src row = gate*HH + wg*16 + e
__global__ void whh_prep(const float* __restrict__ w_hh, unsigned short* __restrict__ dst) {
  int r = blockIdx.x;                 // source row 0..FH-1
  int g = r >> 10;                    // gate 0..3
  int idx = r & (HH - 1);             // hidden element 0..1023
  int wg = idx >> 4;
  int e  = idx & 15;
  size_t drow = (size_t)(wg * 64 + g * 16 + e) * HH;
  size_t srow = (size_t)r * HH;
  for (int k = threadIdx.x; k < HH; k += blockDim.x)
    dst[drow + k] = f2bf_bits(w_hh[srow + k]);
}

// ---------- kernel 1b: plain row-major f32 -> bf16 cast (used for w_ih) ----------
__global__ void cast_rows_bf16(const float* __restrict__ src, unsigned short* __restrict__ dst) {
  size_t base = (size_t)blockIdx.x * HH;
  for (int k = threadIdx.x; k < HH; k += blockDim.x)
    dst[base + k] = f2bf_bits(src[base + k]);
}

// ---------- kernel 1c: fused embedding gather + bf16 cast (only the T used rows) ----------
__global__ void emb_gather_bf16(const int* __restrict__ tokens,
                                const float* __restrict__ embedding,
                                unsigned short* __restrict__ dst) {
  int t = blockIdx.x;                              // 0..TT-1
  size_t srow = (size_t)tokens[t] * HH;
  size_t drow = (size_t)t * HH;
  for (int k = threadIdx.x; k < HH; k += blockDim.x)
    dst[drow + k] = f2bf_bits(embedding[srow + k]);
}

// ---------- kernel 2: x_gates GEMM via WMMA bf16 (pure b128 loads + wmma) ----------
// x_gates[t, n] = sum_k emb_bf[t, k] * wih_bf[n, k]  + (b_ih[n] + b_hh[n])
// Each wave computes one 16x16 C tile; block = 8 waves covering 16(M) x 128(N).
__global__ void xgates_wmma(const unsigned short* __restrict__ emb_bf,
                            const unsigned short* __restrict__ wih_bf,
                            const float* __restrict__ b_ih,
                            const float* __restrict__ b_hh,
                            float* __restrict__ xg) {
  const int lane = threadIdx.x & 31;
  const int wave = threadIdx.x >> 5;
  const int half = lane >> 4;        // 0: lanes 0-15, 1: lanes 16-31
  const int mr   = lane & 15;        // A: row index; B/C: column index

  const int t0 = blockIdx.x * 16;
  const int n0 = (blockIdx.y * 8 + wave) * 16;

  const __bf16* arow = (const __bf16*)(emb_bf + (size_t)(t0 + mr) * HH);
  const __bf16* brow = (const __bf16*)(wih_bf + (size_t)(n0 + mr) * HH);

  v8f acc = {};
  #pragma unroll 4
  for (int k0 = 0; k0 < HH; k0 += 32) {
    // A 16x32 bf16 layout: lanes 0-15 hold K = {k0..k0+7, k0+16..k0+23},
    //                      lanes 16-31 hold K = {k0+8..k0+15, k0+24..k0+31}
    const __bf16* ap = arow + k0 + half * 8;
    v8bf alo = *(const v8bf*)(ap);          // 16B load
    v8bf ahi = *(const v8bf*)(ap + 16);     // 16B load
    v16bf a = __builtin_shufflevector(alo, ahi,
        0, 1, 2, 3, 4, 5, 6, 7, 8, 9, 10, 11, 12, 13, 14, 15);

    // B 32x16 bf16 layout: lanes 0-15 hold K = {k0..k0+7, k0+16..k0+23},
    //                      lanes 16-31 hold K = {k0+8..k0+15, k0+24..k0+31}
    const __bf16* bp = brow + k0 + half * 8;
    v8bf blo = *(const v8bf*)(bp);          // 16B load
    v8bf bhi = *(const v8bf*)(bp + 16);     // 16B load
    v16bf b = __builtin_shufflevector(blo, bhi,
        0, 1, 2, 3, 4, 5, 6, 7, 8, 9, 10, 11, 12, 13, 14, 15);

    acc = __builtin_amdgcn_wmma_f32_16x16x32_bf16(
        /*neg_a=*/false, a, /*neg_b=*/false, b,
        /*c_mod=*/(short)0, acc, /*reuse_a=*/false, /*reuse_b=*/false);
  }

  // C layout: lane -> column (lane&15); VGPR r -> row r + 8*half
  const int n = n0 + mr;
  const float bias = b_ih[n] + b_hh[n];
  #pragma unroll
  for (int r = 0; r < 8; ++r) {
    int row = t0 + r + half * 8;
    xg[(size_t)row * FH + n] = acc[r] + bias;
  }
}

// ---------- kernel 3: persistent LSTM scan (64 WGs, one grid barrier per step) ----------
__global__ void lstm_scan(const float* __restrict__ xg,
                          const unsigned short* __restrict__ whh_bf,
                          float* __restrict__ hbuf,          // [2][HH]
                          float* __restrict__ cfinal,        // [HH]
                          const float* __restrict__ c0,
                          unsigned* __restrict__ counter) {
  __shared__ float h_s[HH];
  __shared__ float gates_s[64];
  __shared__ float c_s[16];

  const int wg   = blockIdx.x;          // 0..63
  const int tid  = threadIdx.x;         // 0..255 (8 waves)
  const int lane = tid & 31;
  const int wave = tid >> 5;
  const int elem_base = wg * 16;
  const unsigned nwg = gridDim.x;

  if (tid < 16) c_s[tid] = c0[elem_base + tid];
  __syncthreads();

  for (int t = 0; t < TT; ++t) {
    const float* cur = hbuf + ((t & 1) * HH);
    float* nxt = hbuf + (((t + 1) & 1) * HH);

    // stage current h into LDS
    for (int j = tid; j < HH; j += 256) h_s[j] = cur[j];
    __syncthreads();

    // each wave: 8 gate rows; dot(w_hh_row(bf16), h) with f32 accumulate
    #pragma unroll
    for (int j = 0; j < 8; ++j) {
      const int lr = wave * 8 + j;             // local row: gate*16 + e
      const int g  = lr >> 4;
      const int e  = lr & 15;
      const unsigned* wrow =
          (const unsigned*)(whh_bf + (size_t)(wg * 64 + lr) * HH);  // 512 bf16 pairs
      float acc = 0.0f;
      for (int p = lane; p < HH / 2; p += 32) {
        unsigned u = wrow[p];
        float2 hv = *(const float2*)&h_s[2 * p];
        acc = fmaf(bfbits2f((unsigned short)(u & 0xffffu)), hv.x, acc);
        acc = fmaf(bfbits2f((unsigned short)(u >> 16)), hv.y, acc);
      }
      acc = wave_reduce_add(acc);
      if (lane == 0) {
        const size_t grow = (size_t)g * HH + elem_base + e;
        gates_s[lr] = acc + xg[(size_t)t * FH + grow];
        if (t + 1 < TT)
          __builtin_prefetch(&xg[(size_t)(t + 1) * FH + grow], 0, 0);
      }
    }
    __syncthreads();

    // cell update for this WG's 16 hidden elements (c never leaves LDS)
    if (tid < 16) {
      float i_g = sigmoidf(gates_s[tid]);
      float f_g = sigmoidf(gates_s[16 + tid]);
      float g_g = tanhf(gates_s[32 + tid]);
      float o_g = sigmoidf(gates_s[48 + tid]);
      float c   = f_g * c_s[tid] + i_g * g_g;
      c_s[tid]  = c;
      nxt[elem_base + tid] = o_g * tanhf(c);
    }
    __threadfence();
    __syncthreads();

    // grid-wide barrier: monotone counter, target = (t+1)*nwg
    if (tid == 0) {
      atomicAdd(counter, 1u);
      const unsigned target = (unsigned)(t + 1) * nwg;
      while (__hip_atomic_load(counter, __ATOMIC_RELAXED,
                               __HIP_MEMORY_SCOPE_AGENT) < target) {
      }
      __threadfence();
    }
    __syncthreads();
  }

  if (tid < 16) cfinal[elem_base + tid] = c_s[tid];
}

// ---------- kernel 4: final projections (one wave per output row) ----------
__global__ void projections(const float* __restrict__ hfin, const float* __restrict__ cfin,
                            const float* __restrict__ Whm, const float* __restrict__ bhm,
                            const float* __restrict__ Whv, const float* __restrict__ bhv,
                            const float* __restrict__ Wcm, const float* __restrict__ bcm,
                            const float* __restrict__ Wcv, const float* __restrict__ bcv,
                            float* __restrict__ out) {
  const int gid  = blockIdx.x * (blockDim.x >> 5) + (threadIdx.x >> 5);
  const int lane = threadIdx.x & 31;
  if (gid >= 4 * LL) return;
  const int sel = gid >> 8;        // 0:hm 1:hv 2:cm 3:cv
  const int row = gid & (LL - 1);
  const float* src = (sel < 2) ? hfin : cfin;
  const float* W; const float* B;
  if (sel == 0)      { W = Whm; B = bhm; }
  else if (sel == 1) { W = Whv; B = bhv; }
  else if (sel == 2) { W = Wcm; B = bcm; }
  else               { W = Wcv; B = bcv; }
  const float* wr = W + (size_t)row * HH;
  float acc = 0.0f;
  for (int k = lane; k < HH; k += 32) acc = fmaf(wr[k], src[k], acc);
  acc = wave_reduce_add(acc);
  if (lane == 0) out[gid] = acc + B[row];
}

// ---------- host ----------
extern "C" void kernel_launch(void* const* d_in, const int* in_sizes, int n_in,
                              void* d_out, int out_size, void* d_ws, size_t ws_size,
                              hipStream_t stream) {
  const int*   tokens    = (const int*)d_in[0];
  const float* h0        = (const float*)d_in[1];
  const float* c0        = (const float*)d_in[2];
  const float* embedding = (const float*)d_in[3];
  const float* w_ih      = (const float*)d_in[4];
  const float* w_hh      = (const float*)d_in[5];
  const float* b_ih      = (const float*)d_in[6];
  const float* b_hh      = (const float*)d_in[7];
  const float* W_hm      = (const float*)d_in[8];
  const float* b_hm      = (const float*)d_in[9];
  const float* W_hv      = (const float*)d_in[10];
  const float* b_hv      = (const float*)d_in[11];
  const float* W_cm      = (const float*)d_in[12];
  const float* b_cm      = (const float*)d_in[13];
  const float* W_cv      = (const float*)d_in[14];
  const float* b_cv      = (const float*)d_in[15];

  char* ws = (char*)d_ws;
  const size_t xg_bytes  = (size_t)TT * FH * sizeof(float);          // 64 MB
  const size_t whh_bytes = (size_t)FH * HH * sizeof(unsigned short); // 8 MB
  const size_t wih_bytes = (size_t)FH * HH * sizeof(unsigned short); // 8 MB
  const size_t emb_bytes = (size_t)TT * HH * sizeof(unsigned short); // 8 MB
  float*          xg      = (float*)ws;
  unsigned short* whh_bf  = (unsigned short*)(ws + xg_bytes);
  unsigned short* wih_bf  = (unsigned short*)(ws + xg_bytes + whh_bytes);
  unsigned short* emb_bf  = (unsigned short*)(ws + xg_bytes + whh_bytes + wih_bytes);
  float*          hbuf    = (float*)(ws + xg_bytes + whh_bytes + wih_bytes + emb_bytes);
  float*          cfin    = hbuf + 2 * HH;                           // HH floats
  unsigned*       counter = (unsigned*)(cfin + HH);

  init_state<<<(HH + 255) / 256, 256, 0, stream>>>(h0, hbuf, counter);
  whh_prep<<<FH, 256, 0, stream>>>(w_hh, whh_bf);
  cast_rows_bf16<<<FH, 256, 0, stream>>>(w_ih, wih_bf);
  emb_gather_bf16<<<TT, 256, 0, stream>>>(tokens, embedding, emb_bf);

  xgates_wmma<<<dim3(TT / 16, FH / 128), 256, 0, stream>>>(
      emb_bf, wih_bf, b_ih, b_hh, xg);

  lstm_scan<<<64, 256, 0, stream>>>(xg, whh_bf, hbuf, cfin, c0, counter);

  // T is even, so the final hidden state lives in buffer (T % 2) == 0
  const float* hfin = hbuf + (TT % 2) * HH;
  projections<<<(4 * LL * 32) / 256, 256, 0, stream>>>(
      hfin, cfin, W_hm, b_hm, W_hv, b_hv, W_cm, b_cm, W_cv, b_cv, (float*)d_out);
}
